// MoELoRALayer_8864812499158
// MI455X (gfx1250) — matrix operand order repaired and checked by reading, not compiled, not run on hardware
//
#include <hip/hip_runtime.h>
#include <hip/hip_bf16.h>

typedef __bf16 bf16;
typedef __attribute__((ext_vector_type(16))) __bf16 v16bf;
typedef __attribute__((ext_vector_type(8)))  __bf16 v8bf;
typedef __attribute__((ext_vector_type(8)))  float   v8f;
typedef __attribute__((ext_vector_type(4)))  unsigned int u32x4;

// ---- problem dims (fixed by the reference) ----
constexpr int BATCH = 2;
constexpr int S_    = 4096;
constexpr int T_    = BATCH * S_;   // 8192 tokens
constexpr int D_    = 4096;
constexpr int O_    = 4096;
constexpr int E_    = 8;
constexpr int R_    = 16;
constexpr int ER_   = E_ * R_;      // 128
constexpr float SCALING_ = 32.0f / 16.0f;

// ---- GEMM tiling ----
constexpr int BM = 128, BN = 128, BK = 64;
constexpr int LSTR = BK + 8;        // 72 halves = 144B row stride: 16B-aligned, conflict-free

// ------------------------------------------------------------------
// fp32 -> bf16 bulk convert (8 elements / thread)
// ------------------------------------------------------------------
__global__ void convert_f32_bf16(const float* __restrict__ src,
                                 bf16* __restrict__ dst, long n) {
  long i = ((long)blockIdx.x * blockDim.x + threadIdx.x) * 8;
  if (i + 8 > n) return;
  const float4* s = reinterpret_cast<const float4*>(src + i);
  float4 a = s[0], b = s[1];
  v8bf o;
  o[0] = (bf16)a.x; o[1] = (bf16)a.y; o[2] = (bf16)a.z; o[3] = (bf16)a.w;
  o[4] = (bf16)b.x; o[5] = (bf16)b.y; o[6] = (bf16)b.z; o[7] = (bf16)b.w;
  *reinterpret_cast<v8bf*>(dst + i) = o;
}

// ------------------------------------------------------------------
// lora_B [E,O,R] fp32 -> Bb [O, E*R] bf16   (Bb[o, e*R+r] = lora_B[e,o,r])
// ------------------------------------------------------------------
__global__ void repack_loraB(const float* __restrict__ lB, bf16* __restrict__ Bbp) {
  int idx = blockIdx.x * 256 + threadIdx.x;     // 0 .. O*E-1
  if (idx >= O_ * E_) return;
  int o = idx >> 3, e = idx & 7;
  const float* src = lB + ((size_t)e * O_ + o) * R_;
  bf16* dst = Bbp + ((size_t)o * ER_ + e * R_);
  v8bf v;
#pragma unroll
  for (int r = 0; r < 8; ++r) v[r] = (bf16)src[r];
  *reinterpret_cast<v8bf*>(dst) = v;
#pragma unroll
  for (int r = 0; r < 8; ++r) v[r] = (bf16)src[8 + r];
  *reinterpret_cast<v8bf*>(dst + 8) = v;
}

// ------------------------------------------------------------------
// Router: one wave32 per token. logits -> softmax -> top2 -> renorm,
// folds in SCALING. Output wsc[T, E] (zero for unselected experts).
// ------------------------------------------------------------------
__global__ void router_kernel(const float* __restrict__ x,
                              const float* __restrict__ Wr,
                              float* __restrict__ wsc) {
  int wave = threadIdx.x >> 5, lane = threadIdx.x & 31;
  int t = blockIdx.x * 8 + wave;
  if (t >= T_) return;
  const float* xt = x + (size_t)t * D_;
  float acc[E_];
#pragma unroll
  for (int e = 0; e < E_; ++e) acc[e] = 0.0f;
  for (int d = lane; d < D_; d += 32) {
    float xv = xt[d];
#pragma unroll
    for (int e = 0; e < E_; ++e) acc[e] += xv * Wr[e * D_ + d];
  }
#pragma unroll
  for (int e = 0; e < E_; ++e)
#pragma unroll
    for (int off = 16; off > 0; off >>= 1)
      acc[e] += __shfl_xor(acc[e], off, 32);
  if (lane == 0) {
    float m = acc[0];
#pragma unroll
    for (int e = 1; e < E_; ++e) m = fmaxf(m, acc[e]);
    float p[E_];
#pragma unroll
    for (int e = 0; e < E_; ++e) p[e] = __expf(acc[e] - m);
    int i1 = 0;
#pragma unroll
    for (int e = 1; e < E_; ++e) if (p[e] > p[i1]) i1 = e;
    int i2 = (i1 == 0) ? 1 : 0;
#pragma unroll
    for (int e = 0; e < E_; ++e) if (e != i1 && p[e] > p[i2]) i2 = e;
    float denom = p[i1] + p[i2];
    float* row = wsc + (size_t)t * E_;
#pragma unroll
    for (int e = 0; e < E_; ++e)
      row[e] = (e == i1 || e == i2) ? (p[e] / denom) * SCALING_ : 0.0f;
  }
}

// ------------------------------------------------------------------
// Register-staged tile: each thread owns 4 x 16B chunks of A and of B.
// chunk layout: row = (tid>>3) + j*32, kc = tid&7  (8 chunks of 16B per row)
// ------------------------------------------------------------------
struct Stage {
  u32x4 a[4];
  u32x4 b[4];
};

__device__ __forceinline__ void load_stage(
    const bf16* __restrict__ Ap, const bf16* __restrict__ Bp, int K,
    int rowA0, int rowB0, int k0, int tid, Stage& st) {
  int r0 = tid >> 3, kc = tid & 7;
#pragma unroll
  for (int j = 0; j < 4; ++j) {
    st.a[j] = *reinterpret_cast<const u32x4*>(
        Ap + (size_t)(rowA0 + r0 + j * 32) * K + k0 + kc * 8);
    st.b[j] = *reinterpret_cast<const u32x4*>(
        Bp + (size_t)(rowB0 + r0 + j * 32) * K + k0 + kc * 8);
  }
}

__device__ __forceinline__ void store_stage(bf16* lA, bf16* lB, int tid,
                                            const Stage& st) {
  int r0 = tid >> 3, kc = tid & 7;
#pragma unroll
  for (int j = 0; j < 4; ++j) {
    *reinterpret_cast<u32x4*>(&lA[(r0 + j * 32) * LSTR + kc * 8]) = st.a[j];
    *reinterpret_cast<u32x4*>(&lB[(r0 + j * 32) * LSTR + kc * 8]) = st.b[j];
  }
}

// ------------------------------------------------------------------
// WMMA K-loop over row-major A[*,K] x row-major B[N,K] (B^T GEMM).
// Block tile BMxBN, 8 waves, each wave a 32x64 sub-tile (2x4 wmma tiles).
// Software-pipelined: next tile's global loads overlap current WMMAs.
// Fragment packing per CDNA5 ISA layouts:
//   A (16x32): lanes<16 hold K{0..7,16..23}; lanes>=16 hold K{8..15,24..31}
//   B (32x16): lanes<16 hold K{0..7,16..23}; lanes>=16 hold K{8..15,24..31}
// ------------------------------------------------------------------
__device__ __forceinline__ void tile_kloop(
    const bf16* __restrict__ Ap, const bf16* __restrict__ Bp, int K,
    int rowA0, int rowB0, bf16* lA, bf16* lB,
    int tid, int l16, int half, int wm, int wn, v8f (&acc)[2][4]) {
  Stage st;
  load_stage(Ap, Bp, K, rowA0, rowB0, 0, tid, st);
  for (int k0 = 0; k0 < K; k0 += BK) {
    __syncthreads();                       // LDS reads of previous tile done
    store_stage(lA, lB, tid, st);
    __syncthreads();
    if (k0 + BK < K)                       // overlap next loads with WMMAs
      load_stage(Ap, Bp, K, rowA0, rowB0, k0 + BK, tid, st);

#pragma unroll
    for (int kk = 0; kk < BK; kk += 32) {
      v16bf afrag[2], bfrag[4];
#pragma unroll
      for (int mi = 0; mi < 2; ++mi) {
        int r = wm + mi * 16 + l16;
        union { v16bf f; v8bf h[2]; } u;
        u.h[0] = *reinterpret_cast<const v8bf*>(&lA[r * LSTR + kk + half * 8]);
        u.h[1] = *reinterpret_cast<const v8bf*>(&lA[r * LSTR + kk + 16 + half * 8]);
        afrag[mi] = u.f;
      }
#pragma unroll
      for (int ni = 0; ni < 4; ++ni) {
        int r = wn + ni * 16 + l16;
        union { v16bf f; v8bf h[2]; } u;
        u.h[0] = *reinterpret_cast<const v8bf*>(&lB[r * LSTR + kk + half * 8]);
        u.h[1] = *reinterpret_cast<const v8bf*>(&lB[r * LSTR + kk + 16 + half * 8]);
        bfrag[ni] = u.f;
      }
#pragma unroll
      for (int mi = 0; mi < 2; ++mi)
#pragma unroll
        for (int ni = 0; ni < 4; ++ni)
          acc[mi][ni] = __builtin_amdgcn_wmma_f32_16x16x32_bf16(
              false, afrag[mi], false, bfrag[ni], (short)0, acc[mi][ni], false, false);
    }
  }
}

// ------------------------------------------------------------------
// h-GEMM: gb[t, e*R+r] = bf16( (xb . Ab^T)[t, e*R+r] * wsc[t,e] )
// M=T, N=128 (one column tile), K=4096
// ------------------------------------------------------------------
__global__ __launch_bounds__(256) void moe_lora_hgemm(
    const bf16* __restrict__ xb, const bf16* __restrict__ Ab,
    const float* __restrict__ wsc, bf16* __restrict__ gb) {
  __shared__ bf16 lA[BM * LSTR];
  __shared__ bf16 lB[BN * LSTR];
  int tid = threadIdx.x;
  int wave = tid >> 5, lane = tid & 31;
  int half = lane >> 4, l16 = lane & 15;
  int blockM = blockIdx.x * BM;
  int wm = (wave & 3) * 32;
  int wn = (wave >> 2) * 64;

  v8f acc[2][4];
  v8f zero = {0.f, 0.f, 0.f, 0.f, 0.f, 0.f, 0.f, 0.f};
#pragma unroll
  for (int mi = 0; mi < 2; ++mi)
#pragma unroll
    for (int ni = 0; ni < 4; ++ni) acc[mi][ni] = zero;

  tile_kloop(xb, Ab, D_, blockM, 0, lA, lB, tid, l16, half, wm, wn, acc);

#pragma unroll
  for (int mi = 0; mi < 2; ++mi)
#pragma unroll
    for (int ni = 0; ni < 4; ++ni) {
      int col = wn + ni * 16 + l16;       // 0..127 == e*R + r
      int e = col >> 4;                   // R == 16
#pragma unroll
      for (int i = 0; i < 8; ++i) {
        int row = blockM + wm + mi * 16 + half * 8 + i;
        float wv = wsc[(size_t)row * E_ + e];
        gb[(size_t)row * ER_ + col] = (bf16)(acc[mi][ni][i] * wv);
      }
    }
}

// ------------------------------------------------------------------
// Main GEMM: out = xb.Wb^T (K=4096)  +  gb.Bb^T (K=128, same accumulators)
//            + bias
// ------------------------------------------------------------------
__global__ __launch_bounds__(256) void moe_lora_main_gemm(
    const bf16* __restrict__ xb, const bf16* __restrict__ Wb,
    const bf16* __restrict__ gb, const bf16* __restrict__ Bbp,
    const float* __restrict__ bias, float* __restrict__ out) {
  __shared__ bf16 lA[BM * LSTR];
  __shared__ bf16 lB[BN * LSTR];
  int tid = threadIdx.x;
  int wave = tid >> 5, lane = tid & 31;
  int half = lane >> 4, l16 = lane & 15;
  int blockM = blockIdx.x * BM;
  int blockN = blockIdx.y * BN;
  int wm = (wave & 3) * 32;
  int wn = (wave >> 2) * 64;

  v8f acc[2][4];
  v8f zero = {0.f, 0.f, 0.f, 0.f, 0.f, 0.f, 0.f, 0.f};
#pragma unroll
  for (int mi = 0; mi < 2; ++mi)
#pragma unroll
    for (int ni = 0; ni < 4; ++ni) acc[mi][ni] = zero;

  // base path: K = 4096
  tile_kloop(xb, Wb, D_, blockM, blockN, lA, lB, tid, l16, half, wm, wn, acc);
  // LoRA second stage folded in: 2 more K-steps over [T,128] x [O,128]^T
  tile_kloop(gb, Bbp, ER_, blockM, blockN, lA, lB, tid, l16, half, wm, wn, acc);

#pragma unroll
  for (int mi = 0; mi < 2; ++mi)
#pragma unroll
    for (int ni = 0; ni < 4; ++ni) {
      int col = blockN + wn + ni * 16 + l16;
      float bv = bias[col];
#pragma unroll
      for (int i = 0; i < 8; ++i) {
        int row = blockM + wm + mi * 16 + half * 8 + i;
        out[(size_t)row * O_ + col] = acc[mi][ni][i] + bv;
      }
    }
}

// ------------------------------------------------------------------
extern "C" void kernel_launch(void* const* d_in, const int* in_sizes, int n_in,
                              void* d_out, int out_size, void* d_ws, size_t ws_size,
                              hipStream_t stream) {
  const float* x        = (const float*)d_in[0];   // [B,S,D]
  const float* W_base   = (const float*)d_in[1];   // [O,D]
  const float* b_base   = (const float*)d_in[2];   // [O]
  const float* W_router = (const float*)d_in[3];   // [E,D]
  const float* lora_A   = (const float*)d_in[4];   // [E,R,D] (== [128, D] flat)
  const float* lora_B   = (const float*)d_in[5];   // [E,O,R]
  float* out = (float*)d_out;                      // [T,O] fp32

  char* ws = (char*)d_ws;
  bf16* xb  = (bf16*)ws;  ws += (size_t)T_ * D_ * sizeof(bf16);   // 64 MB
  bf16* Wb  = (bf16*)ws;  ws += (size_t)O_ * D_ * sizeof(bf16);   // 32 MB
  bf16* Ab  = (bf16*)ws;  ws += (size_t)ER_ * D_ * sizeof(bf16);  // 1 MB
  bf16* gb  = (bf16*)ws;  ws += (size_t)T_ * ER_ * sizeof(bf16);  // 2 MB
  bf16* Bbp = (bf16*)ws;  ws += (size_t)O_ * ER_ * sizeof(bf16);  // 1 MB
  float* wsc = (float*)ws;                                        // 256 KB

  // 1) precision staging
  convert_f32_bf16<<<(T_ * (long)D_) / (8 * 256), 256, 0, stream>>>(x, xb, (long)T_ * D_);
  convert_f32_bf16<<<(O_ * (long)D_) / (8 * 256), 256, 0, stream>>>(W_base, Wb, (long)O_ * D_);
  convert_f32_bf16<<<(ER_ * (long)D_) / (8 * 256), 256, 0, stream>>>(lora_A, Ab, (long)ER_ * D_);
  repack_loraB<<<(O_ * E_) / 256, 256, 0, stream>>>(lora_B, Bbp);

  // 2) router / gating (scaling folded in)
  router_kernel<<<T_ / 8, 256, 0, stream>>>(x, W_router, wsc);

  // 3) LoRA stage 1 (WMMA), epilogue applies gate weights
  moe_lora_hgemm<<<T_ / BM, 256, 0, stream>>>(xb, Ab, wsc, gb);

  // 4) base GEMM + folded LoRA stage 2 + bias (WMMA)
  moe_lora_main_gemm<<<dim3(T_ / BM, O_ / BN), 256, 0, stream>>>(xb, Wb, gb, Bbp, b_base, out);
}
